// MultiHeadedAttention_6055903887423
// MI455X (gfx1250) — matrix operand order repaired and checked by reading, not compiled, not run on hardware
//
#include <hip/hip_runtime.h>

// ---- problem constants (match reference) ----
constexpr int Bc = 4;      // batch
constexpr int Sc = 2048;   // sequence
constexpr int Dc = 1024;   // model dim
constexpr int Hc = 16;     // heads
constexpr int HDc = 64;    // head dim

typedef __attribute__((ext_vector_type(16))) _Float16 v16h;
typedef __attribute__((ext_vector_type(8)))  float    v8f;
typedef __attribute__((ext_vector_type(4)))  unsigned int uint4v;
typedef __attribute__((ext_vector_type(4)))  int       int4v;
typedef __attribute__((ext_vector_type(8)))  int       int8v;

union FragH { v16h v; _Float16 h[16]; uint4v q[2]; };
union FragF { v8f  v; float f[8]; };

// ---------------------------------------------------------------------
// CDNA5 data movers: TDM tensor_load_to_lds (preferred) with fallback to
// per-lane global_load_async_to_lds_b128 (ASYNCcnt-tracked).
// ---------------------------------------------------------------------
#if defined(__HIP_DEVICE_COMPILE__)
#if __has_builtin(__builtin_amdgcn_global_load_async_to_lds_b128)
#define ASYNC_BUILTIN 1
typedef __attribute__((address_space(1))) int4v g_int4;
typedef __attribute__((address_space(3))) int4v l_int4;
#endif
#if __has_builtin(__builtin_amdgcn_tensor_load_to_lds)
#define HAVE_TDM 1
#endif
#endif

__device__ __forceinline__ void async_copy16(const void* g, void* l) {
#if defined(ASYNC_BUILTIN)
  __builtin_amdgcn_global_load_async_to_lds_b128(
      (g_int4*)(size_t)g, (l_int4*)(unsigned)(size_t)l, 0, 0);
#else
  unsigned la = (unsigned)(size_t)l;
  asm volatile("global_load_async_to_lds_b128 %0, %1, off"
               :: "v"(la), "v"(g) : "memory");
#endif
}

__device__ __forceinline__ void wait_async0() {
#if defined(__HIP_DEVICE_COMPILE__) && __has_builtin(__builtin_amdgcn_s_wait_asynccnt)
  __builtin_amdgcn_s_wait_asynccnt(0);
#else
  asm volatile("s_wait_asynccnt 0x0" ::: "memory");
#endif
}

#if defined(HAVE_TDM)
// 2D tile DMA: global (row-major, row stride in elements) -> dense LDS rows.
// f16 elements. D# packed per CDNA5 ISA 8.3/8.4 (groups 2/3 zero => <=2D).
// This toolchain declares the 6-arg builtin:
//   (uint32x4 g0, int32x8 g1, int32x4, int32x4, int32x8, i32 cpol)
__device__ __forceinline__ void tdm_load_2d(const void* gsrc, void* ldst,
                                            unsigned tile_w, unsigned tile_h,
                                            unsigned row_stride_elems) {
  const unsigned long long ga = (unsigned long long)(size_t)gsrc;
  const unsigned lds = (unsigned)(size_t)ldst;
  const unsigned td0 = 0x40000000u;       // tensor_dim0 (OOB bound, huge)
  const unsigned td1 = 0x40000000u;       // tensor_dim1
  const unsigned long long s0 = row_stride_elems;   // tensor_dim0_stride
  const unsigned long long s1 = 0;                  // tensor_dim1_stride (unused 2D)

  uint4v g0;
  g0.x = 1u;                                               // count=1, user mode
  g0.y = lds;                                              // lds_addr (bytes)
  g0.z = (unsigned)(ga & 0xFFFFFFFFu);                     // global_addr[31:0]
  g0.w = (unsigned)((ga >> 32) & 0x01FFFFFFu) | (2u << 30);// addr[56:32] | type=2

  int8v g1;
  g1[0] = (int)(1u << 16);                                 // data_size=1 (2 bytes)
  g1[1] = (int)((td0 & 0xFFFFu) << 16);                    // [63:48] td0 lo16
  g1[2] = (int)(((td0 >> 16) & 0xFFFFu) | ((td1 & 0xFFFFu) << 16));
  g1[3] = (int)(((td1 >> 16) & 0xFFFFu) | ((tile_w & 0xFFFFu) << 16)); // tile_dim0
  g1[4] = (int)(tile_h & 0xFFFFu);                         // tile_dim1 (tile_dim2=0)
  g1[5] = (int)(unsigned)(s0 & 0xFFFFFFFFu);               // td0_stride[31:0]
  g1[6] = (int)(((s0 >> 32) & 0xFFFFu) | ((unsigned)(s1 & 0xFFFFu) << 16));
  g1[7] = (int)(unsigned)((s1 >> 16) & 0xFFFFFFFFu);

  int4v gz4 = (int4v){0, 0, 0, 0};
  int8v gz8 = (int8v){0, 0, 0, 0, 0, 0, 0, 0};
  __builtin_amdgcn_tensor_load_to_lds(g0, g1, gz4, gz4, gz8, 0);
}
#endif  // HAVE_TDM

__device__ __forceinline__ void wait_tiles() {
#if defined(HAVE_TDM)
#if __has_builtin(__builtin_amdgcn_s_wait_tensorcnt)
  __builtin_amdgcn_s_wait_tensorcnt((short)0);
#else
  asm volatile("s_wait_tensorcnt 0x0" ::: "memory");
#endif
#else
  wait_async0();
#endif
}

// =====================================================================
// f32 -> f16 conversion pass (one-time, bandwidth bound)
// =====================================================================
__global__ __launch_bounds__(256)
void cvt_f32_f16(const float* __restrict__ src, _Float16* __restrict__ dst, int n4) {
  int i = blockIdx.x * 256 + threadIdx.x;   // handles 4 elements
  if (i < n4) {
    float4 f = reinterpret_cast<const float4*>(src)[i];
    union { _Float16 h[4]; unsigned long long u; } o;
    o.h[0] = (_Float16)f.x; o.h[1] = (_Float16)f.y;
    o.h[2] = (_Float16)f.z; o.h[3] = (_Float16)f.w;
    reinterpret_cast<unsigned long long*>(dst)[i] = o.u;
  }
}

// =====================================================================
// GEMM: Out[M,N] = A[M,K] @ W[N,K]^T + bias[N]  (A, W already f16)
// 64x64 tile / 128-thread block (4 waves); double-buffered TDM tiles.
// =====================================================================
template <bool OUT_F32>
__global__ __launch_bounds__(128)
void gemm_wmma(const _Float16* __restrict__ A, const _Float16* __restrict__ W,
               const float* __restrict__ bias, void* __restrict__ Out,
               int M, int N, int K) {
  __shared__ alignas(16) _Float16 As[2][64][32];
  __shared__ alignas(16) _Float16 Ws[2][64][32];

  const int tid  = threadIdx.x;
  const int lane = tid & 31;
  const int wv   = tid >> 5;        // 0..3
  const int half = lane >> 4;       // 0/1
  const int lm   = lane & 15;       // 0..15

  const int nbase = blockIdx.x * 64;
  const int mbase = blockIdx.y * 64;

  // fragment gather offsets (16-bit A 16x32 layout; B uses the same k order)
  const int kb0 = half ? 8  : 0;
  const int kb1 = half ? 24 : 16;
  const int kbB = half ? 8 : 0;

  auto issue_tile = [&](int kt, int buf) {
#if defined(HAVE_TDM)
    if (wv == 0) {
      tdm_load_2d(&A[(size_t)mbase * K + kt * 32], &As[buf][0][0], 32, 64, K);
    } else if (wv == 1) {
      tdm_load_2d(&W[(size_t)nbase * K + kt * 32], &Ws[buf][0][0], 32, 64, K);
    }
#else
#pragma unroll
    for (int c = 0; c < 2; ++c) {
      int idx = tid + c * 128;        // 0..255 : 64 rows x 4 chunks(16B)
      int row = idx >> 2;
      int co  = (idx & 3) * 8;        // half offset within 32-wide row
      async_copy16(&A[(size_t)(mbase + row) * K + kt * 32 + co], &As[buf][row][co]);
      async_copy16(&W[(size_t)(nbase + row) * K + kt * 32 + co], &Ws[buf][row][co]);
    }
#endif
  };

  FragF acc[4];
#pragma unroll
  for (int t = 0; t < 4; ++t) acc[t].v = (v8f){0.f,0.f,0.f,0.f,0.f,0.f,0.f,0.f};

  const int nk = K / 32;
  issue_tile(0, 0);
  for (int kt = 0; kt < nk; ++kt) {
    const int buf = kt & 1;
    wait_tiles();         // this wave's DMA for tile kt complete
    __syncthreads();      // all waves' tile-kt writes visible; buf^1 reads done
    if (kt + 1 < nk) issue_tile(kt + 1, buf ^ 1);   // overlap DMA with math

    FragH a;
    const int m = wv * 16 + lm;
    a.q[0] = *reinterpret_cast<const uint4v*>(&As[buf][m][kb0]);
    a.q[1] = *reinterpret_cast<const uint4v*>(&As[buf][m][kb1]);
#pragma unroll
    for (int nt = 0; nt < 4; ++nt) {
      FragH b;
      const _Float16* wrow = &Ws[buf][nt * 16 + lm][kbB];
      b.q[0] = *reinterpret_cast<const uint4v*>(wrow);
      b.q[1] = *reinterpret_cast<const uint4v*>(wrow + 16);
      acc[nt].v = __builtin_amdgcn_wmma_f32_16x16x32_f16(
          false, a.v, false, b.v, (short)0, acc[nt].v, false, false);
    }
  }

  // epilogue: C layout -> row = r + 8*half, col = lm (within 16x16 subtile)
#pragma unroll
  for (int nt = 0; nt < 4; ++nt) {
    const int n  = nbase + nt * 16 + lm;
    const float bv = bias[n];
#pragma unroll
    for (int r = 0; r < 8; ++r) {
      const int mrow = mbase + wv * 16 + r + 8 * half;
      const float val = acc[nt].f[r] + bv;
      if constexpr (OUT_F32) {
        ((float*)Out)[(size_t)mrow * N + n] = val;
      } else {
        ((_Float16*)Out)[(size_t)mrow * N + n] = (_Float16)val;
      }
    }
  }
}

// =====================================================================
// Attention: per block = one (b, h, 64 query rows); flash-style online
// softmax over 32-key tiles. 4 waves, each owns 16 query rows.
// Q/K tiles arrive via TDM (or async DMA); V is transposed manually.
// =====================================================================
__global__ __launch_bounds__(128)
void attn_wmma(const _Float16* __restrict__ qh, const _Float16* __restrict__ kh,
               const _Float16* __restrict__ vh, const int* __restrict__ mask,
               _Float16* __restrict__ outh) {
  __shared__ alignas(16) _Float16 Qs[64][64];      // query tile
  __shared__ alignas(16) _Float16 Ks[32][64];      // key tile (row = key)
  __shared__ alignas(16) _Float16 Vt[64][32];      // value tile transposed [hd][key]
  __shared__ alignas(16) _Float16 Ps[4][16][32];   // per-wave probability tile

  const int tid  = threadIdx.x;
  const int lane = tid & 31;
  const int wv   = tid >> 5;
  const int half = lane >> 4;
  const int lm   = lane & 15;

  const int qbase = blockIdx.x * 64;
  const int h     = blockIdx.y;
  const int b     = blockIdx.z;

  // DMA of Q tile: 64 rows x 64 halves
#if defined(HAVE_TDM)
  if (wv == 0) {
    tdm_load_2d(&qh[((size_t)(b * Sc + qbase)) * Dc + h * HDc], &Qs[0][0],
                64, 64, Dc);
  }
#else
#pragma unroll
  for (int c0 = 0; c0 < 4; ++c0) {
    int c = tid + c0 * 128;
    int row = c >> 3;               // 8 chunks per 128B row
    int co  = (c & 7) * 8;
    async_copy16(&qh[((size_t)(b * Sc + qbase + row)) * Dc + h * HDc + co],
                 &Qs[row][co]);
  }
#endif

  // per-row query mask (reference masks whole query rows)
  int qmask[8];
#pragma unroll
  for (int r = 0; r < 8; ++r) {
    int qrow = qbase + wv * 16 + r + 8 * half;
    qmask[r] = mask[b * Sc + qrow];
  }

  float m_r[8], s_r[8];
  FragF O[4];
#pragma unroll
  for (int r = 0; r < 8; ++r) { m_r[r] = -1e30f; s_r[r] = 0.f; }
#pragma unroll
  for (int t = 0; t < 4; ++t) O[t].v = (v8f){0.f,0.f,0.f,0.f,0.f,0.f,0.f,0.f};

  const float scale = 0.125f;  // 1/sqrt(64)

  for (int kt = 0; kt < Sc / 32; ++kt) {
    __syncthreads();   // previous iteration's consumers are done with Ks/Vt
    // DMA of K tile: 32 rows x 64 halves
#if defined(HAVE_TDM)
    if (wv == 1) {
      tdm_load_2d(&kh[((size_t)(b * Sc + kt * 32)) * Dc + h * HDc], &Ks[0][0],
                  64, 32, Dc);
    }
#else
#pragma unroll
    for (int c0 = 0; c0 < 2; ++c0) {
      int c = tid + c0 * 128;
      int row = c >> 3;
      int co  = (c & 7) * 8;
      async_copy16(&kh[((size_t)(b * Sc + kt * 32 + row)) * Dc + h * HDc + co],
                   &Ks[row][co]);
    }
#endif
    // V tile transposed (overlaps with the DMA above)
    for (int i = tid; i < 32 * 64; i += 128) {
      int row = i >> 6, col = i & 63;
      Vt[col][row] = vh[((size_t)(b * Sc + kt * 32 + row)) * Dc + h * HDc + col];
    }
    wait_tiles();      // covers this wave's Q (first iter) + K DMA
    __syncthreads();

    // ---- scores S = Q @ K^T : 16x32 per wave (2 key subtiles) ----
    FragF sc0, sc1;
    sc0.v = (v8f){0.f,0.f,0.f,0.f,0.f,0.f,0.f,0.f};
    sc1.v = sc0.v;
#pragma unroll
    for (int j = 0; j < 2; ++j) {  // head-dim chunks of 32
      FragH a;
      const int m = wv * 16 + lm;
      const int kb0 = 32 * j + (half ? 8 : 0);
      const int kb1 = 32 * j + (half ? 24 : 16);
      a.q[0] = *reinterpret_cast<const uint4v*>(&Qs[m][kb0]);
      a.q[1] = *reinterpret_cast<const uint4v*>(&Qs[m][kb1]);
      const int kbB = 32 * j + (half ? 8 : 0);
      {
        FragH bk;
        const _Float16* kr = &Ks[lm][kbB];  // keys 0..15
        bk.q[0] = *reinterpret_cast<const uint4v*>(kr);
        bk.q[1] = *reinterpret_cast<const uint4v*>(kr + 16);
        sc0.v = __builtin_amdgcn_wmma_f32_16x16x32_f16(
            false, a.v, false, bk.v, (short)0, sc0.v, false, false);
      }
      {
        FragH bk;
        const _Float16* kr = &Ks[16 + lm][kbB];  // keys 16..31
        bk.q[0] = *reinterpret_cast<const uint4v*>(kr);
        bk.q[1] = *reinterpret_cast<const uint4v*>(kr + 16);
        sc1.v = __builtin_amdgcn_wmma_f32_16x16x32_f16(
            false, a.v, false, bk.v, (short)0, sc1.v, false, false);
      }
    }

    // ---- online softmax update per row (C layout: VGPR r <-> row r+8*half) ----
#pragma unroll
    for (int r = 0; r < 8; ++r) {
      float x0 = qmask[r] ? sc0.f[r] * scale : -1e9f;
      float x1 = qmask[r] ? sc1.f[r] * scale : -1e9f;
      float mx = fmaxf(x0, x1);
#pragma unroll
      for (int off = 1; off < 16; off <<= 1)
        mx = fmaxf(mx, __shfl_xor(mx, off, 32));
      float mnew  = fmaxf(m_r[r], mx);
      float alpha = __expf(m_r[r] - mnew);
      float e0 = __expf(x0 - mnew);
      float e1 = __expf(x1 - mnew);
      float rs = e0 + e1;
#pragma unroll
      for (int off = 1; off < 16; off <<= 1)
        rs += __shfl_xor(rs, off, 32);
      s_r[r] = s_r[r] * alpha + rs;
      m_r[r] = mnew;
#pragma unroll
      for (int t = 0; t < 4; ++t) O[t].f[r] *= alpha;
      Ps[wv][r + 8 * half][lm]      = (_Float16)e0;
      Ps[wv][r + 8 * half][lm + 16] = (_Float16)e1;
    }
    // same-wave LDS ops are in-order; only this wave touches Ps[wv]

    // ---- O += P @ V : K-dim = 32 keys, N = 64 head-dims (4 subtiles) ----
    FragH ap;
    {
      const int kb0 = half ? 8 : 0;
      const int kb1 = half ? 24 : 16;
      ap.q[0] = *reinterpret_cast<const uint4v*>(&Ps[wv][lm][kb0]);
      ap.q[1] = *reinterpret_cast<const uint4v*>(&Ps[wv][lm][kb1]);
    }
    const int kbV = half ? 8 : 0;
#pragma unroll
    for (int t = 0; t < 4; ++t) {
      FragH bv;
      const _Float16* vr = &Vt[t * 16 + lm][kbV];
      bv.q[0] = *reinterpret_cast<const uint4v*>(vr);
      bv.q[1] = *reinterpret_cast<const uint4v*>(vr + 16);
      O[t].v = __builtin_amdgcn_wmma_f32_16x16x32_f16(
          false, ap.v, false, bv.v, (short)0, O[t].v, false, false);
    }
  }

  // ---- normalize and store (f16 intermediate for final projection) ----
#pragma unroll
  for (int t = 0; t < 4; ++t) {
#pragma unroll
    for (int r = 0; r < 8; ++r) {
      const int qrow = qbase + wv * 16 + r + 8 * half;
      const int col  = h * HDc + t * 16 + lm;
      float val = O[t].f[r] / s_r[r];
      outh[((size_t)(b * Sc + qrow)) * Dc + col] = (_Float16)val;
    }
  }
}

// =====================================================================
extern "C" void kernel_launch(void* const* d_in, const int* in_sizes, int n_in,
                              void* d_out, int out_size, void* d_ws, size_t ws_size,
                              hipStream_t stream) {
  const float* q    = (const float*)d_in[0];
  const float* k    = (const float*)d_in[1];
  const float* v    = (const float*)d_in[2];
  const float* Wq   = (const float*)d_in[3];
  const float* bq   = (const float*)d_in[4];
  const float* Wk   = (const float*)d_in[5];
  const float* bk   = (const float*)d_in[6];
  const float* Wv   = (const float*)d_in[7];
  const float* bv   = (const float*)d_in[8];
  const float* Wo   = (const float*)d_in[9];
  const float* bo   = (const float*)d_in[10];
  const int*   mask = (const int*)d_in[11];
  float* out = (float*)d_out;

  const int M = Bc * Sc;           // 8192
  const int N = Dc, K = Dc;        // 1024

  // workspace layout (aliased, sequential dependencies on one stream):
  //   x16 : 16 MB f16 staging (q16/k16/v16 sequentially; later attn output)
  //   w16 :  2 MB f16 weight staging (Wq/Wk/Wv/Wo sequentially)
  //   qh/kh/vh : 16 MB each, f16 projected tensors
  const size_t TSZ = (size_t)M * Dc * sizeof(_Float16);   // 16 MB
  const size_t WSZ = (size_t)Dc * Dc * sizeof(_Float16);  //  2 MB
  char* ws = (char*)d_ws;
  _Float16* x16 = (_Float16*)(ws);
  _Float16* w16 = (_Float16*)(ws + TSZ);
  _Float16* qh  = (_Float16*)(ws + TSZ + WSZ);
  _Float16* kh  = (_Float16*)(ws + 2 * TSZ + WSZ);
  _Float16* vh  = (_Float16*)(ws + 3 * TSZ + WSZ);
  _Float16* hid = x16;   // free after the V projection

  const int n4_in = M * Dc / 4;       // 2M  vec4 elements
  const int n4_w  = Dc * Dc / 4;      // 256K vec4 elements
  dim3 bCvt(256);
  dim3 gCvtIn((n4_in + 255) / 256);
  dim3 gCvtW((n4_w + 255) / 256);

  dim3 gGemm(N / 64, M / 64);         // (16, 128)
  dim3 bGemm(128);

  // Q projection
  cvt_f32_f16<<<gCvtIn, bCvt, 0, stream>>>(q, x16, n4_in);
  cvt_f32_f16<<<gCvtW,  bCvt, 0, stream>>>(Wq, w16, n4_w);
  gemm_wmma<false><<<gGemm, bGemm, 0, stream>>>(x16, w16, bq, qh, M, N, K);
  // K projection
  cvt_f32_f16<<<gCvtIn, bCvt, 0, stream>>>(k, x16, n4_in);
  cvt_f32_f16<<<gCvtW,  bCvt, 0, stream>>>(Wk, w16, n4_w);
  gemm_wmma<false><<<gGemm, bGemm, 0, stream>>>(x16, w16, bk, kh, M, N, K);
  // V projection
  cvt_f32_f16<<<gCvtIn, bCvt, 0, stream>>>(v, x16, n4_in);
  cvt_f32_f16<<<gCvtW,  bCvt, 0, stream>>>(Wv, w16, n4_w);
  gemm_wmma<false><<<gGemm, bGemm, 0, stream>>>(x16, w16, bv, vh, M, N, K);

  // attention (writes hid == x16, which is free now)
  dim3 gAttn(Sc / 64, Hc, Bc);        // (32, 16, 4)
  attn_wmma<<<gAttn, dim3(128), 0, stream>>>(qh, kh, vh, mask, hid);

  // output projection (f16 in, f32 out to d_out)
  cvt_f32_f16<<<gCvtW, bCvt, 0, stream>>>(Wo, w16, n4_w);
  gemm_wmma<true><<<gGemm, bGemm, 0, stream>>>(hid, w16, bo, out, M, N, K);
}
